// MoEBlock_64733747085415
// MI455X (gfx1250) — matrix operand order repaired and checked by reading, not compiled, not run on hardware
//
#include <hip/hip_runtime.h>
#include <hip/hip_bf16.h>
#include <math.h>

// ---------------------------------------------------------------------------
// MoE block for MI455X (gfx1250): routed top-2 experts + shared expert,
// bf16 WMMA (v_wmma_f32_16x16x32_bf16) GEMMs with fp32 accumulation.
// 64x64 block tiles, 4 waves/block, 2x2 WMMA accumulators per wave.
// LDS tiles staged by the Tensor Data Mover (tensor_load_to_lds / TENSORcnt).
// ---------------------------------------------------------------------------

typedef __bf16 bf16_t;
typedef __attribute__((ext_vector_type(16))) __bf16 v16bf;
typedef __attribute__((ext_vector_type(8)))  __bf16 v8bf;
typedef __attribute__((ext_vector_type(4)))  __bf16 v4bf;
typedef __attribute__((ext_vector_type(8)))  float  v8f;
typedef __attribute__((ext_vector_type(4)))  unsigned int u32x4;
typedef __attribute__((ext_vector_type(8)))  int          i32x8;
typedef __attribute__((ext_vector_type(4)))  int          i32x4;

static constexpr int kB   = 2;
static constexpr int kS   = 2048;
static constexpr int kD   = 1024;
static constexpr int kI   = 4096;
static constexpr int kE   = 8;
static constexpr int kTOP = 2;
static constexpr int kN   = kB * kS;            // 4096 tokens

static constexpr int BM = 64, BN = 64, BK = 32; // block tile sizes
static constexpr int LDSPAD = 8;                // bf16 pad -> 80B LDS row stride
static constexpr int MAXROWS = kN * kTOP + kE * BM; // 8704, BM-multiple

// ---------------------------------------------------------------------------
// TDM: issue a 2D tile load (rows x tile_k bf16 elems, row stride in elems)
// into LDS with 16B padding after each 64B row (matches [rows][BK+8] layout).
// Descriptor packing per CDNA5 ISA section 8.3/8.4 (D# groups 0/1; 2D => g2=g3=0).
// This toolchain exposes the 6-arg builtin (extra int32x8 group, then cpol).
// ---------------------------------------------------------------------------
__device__ __forceinline__ void tdm_load_tile(unsigned lds_addr,
                                              const bf16_t* gsrc,
                                              int rows, int tile_k,
                                              long long row_stride_elems) {
    unsigned long long ga = (unsigned long long)(size_t)gsrc;
    u32x4 g0;
    g0[0] = 1u;                                        // count=1 (valid), user mode
    g0[1] = lds_addr;                                  // LDS byte address
    g0[2] = (unsigned)(ga & 0xffffffffull);            // global addr [31:0]
    g0[3] = (unsigned)((ga >> 32) & 0x1ffffffull)      // global addr [56:32]
          | (2u << 30);                                // type = 2 ("image")

    unsigned long long dim0   = (unsigned long long)(unsigned)tile_k;
    unsigned long long dim1   = (unsigned long long)(unsigned)rows;
    unsigned long long stride = (unsigned long long)row_stride_elems;

    i32x8 g1;
    g1[0] = (int)((1u << 16)      // data_size = 1 -> 2 bytes
                | (1u << 20)      // pad_enable
                | (3u << 22)      // pad_interval code 3 -> every 16 DWORDs (64B)
                | (3u << 25));    // pad_amount   code 3 -> 4 DWORDs (16B)
    g1[1] = (int)((unsigned)((dim0 & 0xffffull) << 16));            // [47:32]=0, [63:48]=dim0.lo
    g1[2] = (int)((unsigned)((dim0 >> 16) & 0xffffull)              // [79:64]=dim0.hi
                | (unsigned)((dim1 & 0xffffull) << 16));            // [95:80]=dim1.lo
    g1[3] = (int)((unsigned)((dim1 >> 16) & 0xffffull)              // [111:96]=dim1.hi
                | ((unsigned)tile_k << 16));                        // [127:112]=tile_dim0
    g1[4] = (int)((unsigned)rows & 0xffffu);                        // [143:128]=tile_dim1
    g1[5] = (int)(unsigned)(stride & 0xffffffffull);                // [191:160]=stride.lo
    g1[6] = (int)(unsigned)((stride >> 32) & 0xffffull);            // [207:192]=stride.hi
    g1[7] = 0;

    i32x4 gz4 = {};
    i32x8 gz8 = {};
    __builtin_amdgcn_tensor_load_to_lds(g0, g1, gz4, gz4, gz8, 0);
}

// ---------------------------------------------------------------------------
// fp32 -> bf16 conversion (vectorized x4)
// ---------------------------------------------------------------------------
__global__ __launch_bounds__(256)
void cvt_kernel(const float* __restrict__ src, bf16_t* __restrict__ dst, int n4) {
    int i = blockIdx.x * blockDim.x + threadIdx.x;
    if (i >= n4) return;
    float4 v = ((const float4*)src)[i];
    v4bf o;
    o[0] = (__bf16)v.x; o[1] = (__bf16)v.y; o[2] = (__bf16)v.z; o[3] = (__bf16)v.w;
    ((v4bf*)dst)[i] = o;
}

// ---------------------------------------------------------------------------
// init: zero per-expert counts, mark all routed rows invalid
// ---------------------------------------------------------------------------
__global__ __launch_bounds__(256)
void init_kernel(int* __restrict__ counts, int* __restrict__ row_token) {
    int i = blockIdx.x * blockDim.x + threadIdx.x;
    if (i < kE) counts[i] = 0;
    if (i < MAXROWS) row_token[i] = -1;
}

// ---------------------------------------------------------------------------
// gate: one wave32 per token. logits -> softmax -> top-2 (prob weights)
// ---------------------------------------------------------------------------
__global__ __launch_bounds__(256)
void gate_kernel(const float* __restrict__ x, const float* __restrict__ gw,
                 int* __restrict__ tk_idx, float* __restrict__ tk_w,
                 int* __restrict__ counts) {
    int gid  = blockIdx.x * blockDim.x + threadIdx.x;
    int tok  = gid >> 5;
    int lane = gid & 31;
    if (tok >= kN) return;

    const float* xr = x + (size_t)tok * kD;
    float acc[kE];
#pragma unroll
    for (int e = 0; e < kE; e++) acc[e] = 0.f;

    for (int d = lane; d < kD; d += 32) {
        float xv = xr[d];
#pragma unroll
        for (int e = 0; e < kE; e++) acc[e] += xv * gw[e * kD + d];
    }
#pragma unroll
    for (int e = 0; e < kE; e++) {
#pragma unroll
        for (int off = 16; off > 0; off >>= 1)
            acc[e] += __shfl_xor(acc[e], off, 32);
    }

    if (lane == 0) {
        float mx = acc[0];
#pragma unroll
        for (int e = 1; e < kE; e++) mx = fmaxf(mx, acc[e]);
        float p[kE], s = 0.f;
#pragma unroll
        for (int e = 0; e < kE; e++) { p[e] = expf(acc[e] - mx); s += p[e]; }
        float inv = 1.f / s;
        int   i0 = 0; float v0 = -1.f;
#pragma unroll
        for (int e = 0; e < kE; e++) {
            p[e] *= inv;
            if (p[e] > v0) { v0 = p[e]; i0 = e; }
        }
        int i1 = 0; float v1 = -1.f;
#pragma unroll
        for (int e = 0; e < kE; e++) {
            if (e != i0 && p[e] > v1) { v1 = p[e]; i1 = e; }
        }
        tk_idx[tok * 2 + 0] = i0; tk_w[tok * 2 + 0] = v0;
        tk_idx[tok * 2 + 1] = i1; tk_w[tok * 2 + 1] = v1;
        atomicAdd(&counts[i0], 1);
        atomicAdd(&counts[i1], 1);
    }
}

// ---------------------------------------------------------------------------
// prefix: padded (BM-aligned) per-expert segment offsets; reset cursors
// ---------------------------------------------------------------------------
__global__ void prefix_kernel(const int* __restrict__ counts,
                              int* __restrict__ offsets,
                              int* __restrict__ cursor) {
    if (blockIdx.x == 0 && threadIdx.x == 0) {
        int acc = 0;
        for (int e = 0; e < kE; e++) {
            offsets[e] = acc;
            acc += (counts[e] + BM - 1) / BM * BM;
            cursor[e] = 0;
        }
        offsets[kE] = acc;  // padded total (multiple of BM)
    }
}

// ---------------------------------------------------------------------------
// scatter tokens into expert segments
// ---------------------------------------------------------------------------
__global__ __launch_bounds__(256)
void scatter_kernel(const int* __restrict__ tk_idx, const float* __restrict__ tk_w,
                    const int* __restrict__ offsets, int* __restrict__ cursor,
                    int* __restrict__ row_token, float* __restrict__ row_w) {
    int tok = blockIdx.x * blockDim.x + threadIdx.x;
    if (tok >= kN) return;
#pragma unroll
    for (int k = 0; k < kTOP; k++) {
        int   e = tk_idx[tok * kTOP + k];
        float w = tk_w[tok * kTOP + k];
        int slot = atomicAdd(&cursor[e], 1);
        int pos  = offsets[e] + slot;
        row_token[pos] = tok;
        row_w[pos]     = w;
    }
}

__device__ __forceinline__ float gelu_exact(float v) {
    return 0.5f * v * (1.0f + erff(v * 0.7071067811865475f));
}

// ---------------------------------------------------------------------------
// Tiled bf16 WMMA GEMM. STAGE 1: h = gelu(A @ W^T + b)  (store bf16)
//                       STAGE 2: out(+)= (A @ W^T + b)  (routed: weighted atomic)
// Block: 128 threads = 4 waves in a 2x2 grid; each wave owns a 32x32 output
// patch = 2x2 WMMA accumulators, reusing its 2 A-frags x 2 B-frags per K-step.
// Tiles staged into LDS by the TDM (wave0: W tile, wave1: X tile); the gathered
// routed-stage1 X tile uses a manual per-row copy instead.
// ---------------------------------------------------------------------------
template<int STAGE, bool ROUTED>
__global__ __launch_bounds__(128)
void ffn_kernel(const bf16_t* __restrict__ Abase,
                const bf16_t* __restrict__ Wbase,
                const float*  __restrict__ biasBase,
                bf16_t* __restrict__ h_out,
                float*  __restrict__ out,
                const int*   __restrict__ offsets,
                const int*   __restrict__ row_token,
                const float* __restrict__ row_w) {
    constexpr int Kdim  = (STAGE == 1) ? kD : kI;
    constexpr int Ncols = (STAGE == 1) ? kI : kD;
    constexpr bool GATHER = (STAGE == 1 && ROUTED);

    const int row0    = blockIdx.y * BM;
    const int colBase = blockIdx.x * BN;

    int e = 0;
    if constexpr (ROUTED) {
        if (row0 >= offsets[kE]) return;   // beyond padded total: uniform exit
        while (e < kE - 1 && row0 >= offsets[e + 1]) e++;
    }
    const bf16_t* W    = Wbase + (size_t)e * ((size_t)kI * kD);
    const float*  bias = biasBase + (size_t)e * Ncols;

    __shared__ bf16_t Xs [BM][BK + LDSPAD];   // 64 x 40 bf16 (80B row stride)
    __shared__ bf16_t Wsh[BN][BK + LDSPAD];

    const int t     = threadIdx.x;
    const int waveId = t >> 5;
    const int lrow  = t & 63;            // manual LDS fill: row (0..63)
    const int lcol  = (t >> 6) * 16;     // manual LDS fill: column half (0/16)

    // Gathered row source (routed stage 1 only)
    const bf16_t* arow = nullptr;
    if constexpr (GATHER) {
        int tok = row_token[row0 + lrow];
        arow = (tok >= 0) ? (Abase + (size_t)tok * Kdim) : nullptr;
    }
    const bf16_t* wtile = W + (size_t)colBase * Kdim;
    const bf16_t* xtile = Abase + (size_t)row0 * Kdim;

    const unsigned ldsW = (unsigned)(size_t)&Wsh[0][0];
    const unsigned ldsX = (unsigned)(size_t)&Xs[0][0];

    const int lane  = t & 31;
    const int m     = lane & 15;
    const int hi    = lane >> 4;         // lane group 0/1
    const int waveM = waveId >> 1;       // 0..1
    const int waveN = waveId & 1;        // 0..1

    v8f c[2][2] = {};

    for (int kb = 0; kb < Kdim; kb += BK) {
        // ---- stage tiles into LDS ----
        if (waveId == 0)
            tdm_load_tile(ldsW, wtile + kb, BN, BK, Kdim);
        if constexpr (GATHER) {
            v8bf xv0 = {}, xv1 = {};
            if (arow) {
                xv0 = *(const v8bf*)(arow + kb + lcol);
                xv1 = *(const v8bf*)(arow + kb + lcol + 8);
                __builtin_prefetch(arow + kb + BK, 0, 1);   // global_prefetch_b8
            }
            *(v8bf*)&Xs[lrow][lcol]     = xv0;
            *(v8bf*)&Xs[lrow][lcol + 8] = xv1;
        } else {
            if (waveId == 1)
                tdm_load_tile(ldsX, xtile + kb, BM, BK, Kdim);
        }
        __builtin_amdgcn_s_wait_tensorcnt(0);
        __syncthreads();

        // ---- build WMMA fragments per ISA 7.12.2 layouts ----
        // A (16x32, M=m): lo lanes K=0..7 & 16..23, hi lanes K=8..15 & 24..31
        union bfu { v16bf v; v8bf h[2]; };
        bfu Af[2], Bf[2];
#pragma unroll
        for (int i = 0; i < 2; i++) {
            const bf16_t* ap = &Xs[waveM * 32 + i * 16 + m][hi * 8];
            Af[i].h[0] = *(const v8bf*)ap;
            Af[i].h[1] = *(const v8bf*)(ap + 16);
            // B (32x16, N=m): lo lanes K=0..7 & 16..23, hi lanes K=8..15 & 24..31 (B = W^T)
            const bf16_t* bp = &Wsh[waveN * 32 + i * 16 + m][hi * 8];
            Bf[i].h[0] = *(const v8bf*)bp;
            Bf[i].h[1] = *(const v8bf*)(bp + 16);
        }
#pragma unroll
        for (int i = 0; i < 2; i++)
#pragma unroll
            for (int j = 0; j < 2; j++)
                c[i][j] = __builtin_amdgcn_wmma_f32_16x16x32_bf16(
                    false, Af[i].v, false, Bf[j].v, (short)0, c[i][j], false, false);
        __syncthreads();
    }

    // ---- epilogue: C layout is M = r + 8*hi (VGPR r), N = m ----
#pragma unroll
    for (int j = 0; j < 2; j++) {
        const int gc = colBase + waveN * 32 + j * 16 + m;
        const float bv = bias[gc];
#pragma unroll
        for (int i = 0; i < 2; i++) {
#pragma unroll
            for (int r = 0; r < 8; r++) {
                const int gr = row0 + waveM * 32 + i * 16 + r + hi * 8;
                float v = c[i][j][r] + bv;
                if constexpr (STAGE == 1) {
                    h_out[(size_t)gr * kI + gc] = (bf16_t)gelu_exact(v);
                } else {
                    if constexpr (ROUTED) {
                        int tok = row_token[gr];
                        if (tok >= 0)
                            atomicAdd(&out[(size_t)tok * kD + gc], row_w[gr] * v);
                    } else {
                        out[(size_t)gr * kD + gc] = v;   // shared path: base value
                    }
                }
            }
        }
    }
}

// ---------------------------------------------------------------------------
// host launcher
// ---------------------------------------------------------------------------
extern "C" void kernel_launch(void* const* d_in, const int* in_sizes, int n_in,
                              void* d_out, int out_size, void* d_ws, size_t ws_size,
                              hipStream_t stream) {
    (void)in_sizes; (void)n_in; (void)out_size; (void)ws_size;

    const float* x      = (const float*)d_in[0];
    const float* gate_w = (const float*)d_in[1];
    const float* e_w1   = (const float*)d_in[2];
    const float* e_b1   = (const float*)d_in[3];
    const float* e_w2   = (const float*)d_in[4];
    const float* e_b2   = (const float*)d_in[5];
    const float* s_w1   = (const float*)d_in[6];
    const float* s_b1   = (const float*)d_in[7];
    const float* s_w2   = (const float*)d_in[8];
    const float* s_b2   = (const float*)d_in[9];
    float* out = (float*)d_out;

    // workspace carve-out (256B aligned)
    char* ws = (char*)d_ws;
    size_t off = 0;
    auto carve = [&](size_t bytes) -> void* {
        void* p = ws + off;
        off += (bytes + 255) & ~(size_t)255;
        return p;
    };
    bf16_t* xbf   = (bf16_t*)carve((size_t)kN * kD * 2);
    bf16_t* w1bf  = (bf16_t*)carve((size_t)kE * kI * kD * 2);
    bf16_t* w2bf  = (bf16_t*)carve((size_t)kE * kD * kI * 2);
    bf16_t* sw1bf = (bf16_t*)carve((size_t)kI * kD * 2);
    bf16_t* sw2bf = (bf16_t*)carve((size_t)kD * kI * 2);
    bf16_t* hbuf  = (bf16_t*)carve((size_t)MAXROWS * kI * 2); // reused shared+routed
    int*   counts    = (int*)carve(kE * 4);
    int*   offsets   = (int*)carve((kE + 1) * 4);
    int*   cursor    = (int*)carve(kE * 4);
    int*   row_token = (int*)carve(MAXROWS * 4);
    float* row_w     = (float*)carve(MAXROWS * 4);
    int*   tk_idx    = (int*)carve((size_t)kN * kTOP * 4);
    float* tk_w      = (float*)carve((size_t)kN * kTOP * 4);

    // 1) fp32 -> bf16 conversions (bf16 weights ~134MB -> resident in 192MB L2)
    auto cvt = [&](const float* s, bf16_t* d, size_t n) {
        int n4 = (int)(n / 4);
        cvt_kernel<<<(n4 + 255) / 256, 256, 0, stream>>>(s, d, n4);
    };
    cvt(x,    xbf,   (size_t)kN * kD);
    cvt(e_w1, w1bf,  (size_t)kE * kI * kD);
    cvt(e_w2, w2bf,  (size_t)kE * kD * kI);
    cvt(s_w1, sw1bf, (size_t)kI * kD);
    cvt(s_w2, sw2bf, (size_t)kD * kI);

    // 2) routing
    init_kernel<<<(MAXROWS + 255) / 256, 256, 0, stream>>>(counts, row_token);
    gate_kernel<<<(kN * 32) / 256, 256, 0, stream>>>(x, gate_w, tk_idx, tk_w, counts);
    prefix_kernel<<<1, 1, 0, stream>>>(counts, offsets, cursor);
    scatter_kernel<<<(kN + 255) / 256, 256, 0, stream>>>(tk_idx, tk_w, offsets, cursor,
                                                         row_token, row_w);

    // 3) shared expert: h = gelu(x@W1^T+b1); out = h@W2^T+b2  (writes base output)
    {
        dim3 g1(kI / BN, kN / BM);   // (64, 64)
        ffn_kernel<1, false><<<g1, 128, 0, stream>>>(xbf, sw1bf, s_b1, hbuf, nullptr,
                                                     nullptr, nullptr, nullptr);
        dim3 g2(kD / BN, kN / BM);   // (16, 64)
        ffn_kernel<2, false><<<g2, 128, 0, stream>>>(hbuf, sw2bf, s_b2, nullptr, out,
                                                     nullptr, nullptr, nullptr);
    }

    // 4) routed experts: per-segment GEMMs; weighted atomic accumulate into out
    {
        dim3 g1(kI / BN, MAXROWS / BM);   // (64, 136)
        ffn_kernel<1, true><<<g1, 128, 0, stream>>>(xbf, w1bf, e_b1, hbuf, nullptr,
                                                    offsets, row_token, row_w);
        dim3 g2(kD / BN, MAXROWS / BM);   // (16, 136)
        ffn_kernel<2, true><<<g2, 128, 0, stream>>>(hbuf, w2bf, e_b2, nullptr, out,
                                                    offsets, row_token, row_w);
    }
}
